// Spatial_dot_80951543595350
// MI455X (gfx1250) — matrix operand order repaired and checked by reading, not compiled, not run on hardware
//
#include <hip/hip_runtime.h>
#include <hip/hip_bf16.h>

typedef __attribute__((ext_vector_type(16))) _Float16 v16h;
typedef __attribute__((ext_vector_type(8)))  _Float16 v8h;
typedef __attribute__((ext_vector_type(8)))  float    v8f;

#define NBT 32      // B*T slices
#define NC  64      // channels
#define NN  1024    // H*W
#define NF  10      // num_feat
#define NFP 16      // F padded for WMMA

// ---------------------------------------------------------------------------
// Kernel 1: three 1x1-conv projections.  Q,K stored f16 n-major [n][16]
// (zero-padded f=10..15) so a WMMA lane can grab 8 contiguous halves.
// G stored f16 f-major [16][1024] to serve as B-operand of the out-GEMM.
// ---------------------------------------------------------------------------
__global__ void proj_kernel(const float* __restrict__ x1,
                            const float* __restrict__ x2,
                            const float* __restrict__ W1, const float* __restrict__ b1,
                            const float* __restrict__ W2, const float* __restrict__ b2,
                            const float* __restrict__ W3, const float* __restrict__ b3,
                            _Float16* __restrict__ Qh, _Float16* __restrict__ Kh,
                            _Float16* __restrict__ Gh)
{
    int bt = blockIdx.x;
    int n  = blockIdx.y * blockDim.x + threadIdx.x;
    const float* x1s = x1 + (size_t)bt * NC * NN;
    const float* x2s = x2 + (size_t)bt * NC * NN;

    float q[NF], k[NF], g[NF];
#pragma unroll
    for (int f = 0; f < NF; ++f) { q[f] = b1[f]; k[f] = b2[f]; g[f] = b3[f]; }

    for (int c = 0; c < NC; ++c) {
        float a1 = x1s[c * NN + n];
        float a2 = x2s[c * NN + n];
#pragma unroll
        for (int f = 0; f < NF; ++f) {
            q[f] = fmaf(W1[f * NC + c], a1, q[f]);
            k[f] = fmaf(W2[f * NC + c], a2, k[f]);
            g[f] = fmaf(W3[f * NC + c], a1, g[f]);
        }
    }
    _Float16* Qs = Qh + (size_t)bt * NN * NFP;
    _Float16* Ks = Kh + (size_t)bt * NN * NFP;
    _Float16* Gs = Gh + (size_t)bt * NFP * NN;
#pragma unroll
    for (int f = 0; f < NFP; ++f) {
        Qs[n * NFP + f] = (f < NF) ? (_Float16)q[f] : (_Float16)0.f;
        Ks[n * NFP + f] = (f < NF) ? (_Float16)k[f] : (_Float16)0.f;
        Gs[f * NN + n]  = (f < NF) ? (_Float16)g[f] : (_Float16)0.f;
    }
}

// ---------------------------------------------------------------------------
// Kernel 2: per-column softmax stats.  softmax axis is n (query axis), so for
// every column m we need max_n S[n,m] and sum_n exp(S-max).  Online update,
// one thread per m, streaming over n (S recomputed from f16 Q/K so the stats
// match the WMMA-recomputed S exactly).
// ---------------------------------------------------------------------------
__global__ void colstats_kernel(const _Float16* __restrict__ Qh,
                                const _Float16* __restrict__ Kh,
                                float* __restrict__ colmax,
                                float* __restrict__ colsum)
{
    int bt = blockIdx.x;
    int m  = blockIdx.y * blockDim.x + threadIdx.x;
    const _Float16* Qs = Qh + (size_t)bt * NN * NFP;
    const _Float16* Ks = Kh + (size_t)bt * NN * NFP;

    float kc[NF];
#pragma unroll
    for (int f = 0; f < NF; ++f) kc[f] = (float)Ks[m * NFP + f];

    float M = -3.0e38f, S = 0.f;
    for (int n = 0; n < NN; ++n) {
        const _Float16* qp = Qs + n * NFP;
        float s = 0.f;
#pragma unroll
        for (int f = 0; f < NF; ++f) s = fmaf((float)qp[f], kc[f], s);
        if (s <= M) {
            S += __expf(s - M);
        } else {
            S = S * __expf(M - s) + 1.f;
            M = s;
        }
    }
    colmax[bt * NN + m] = M;
    colsum[bt * NN + m] = S;
}

// ---------------------------------------------------------------------------
// Kernel 3: fused attention+output.  One wave32 owns 16 n-rows of one slice.
// Per 32-m chunk: 2x v_wmma_f32_16x16x32_f16 to get S tiles, activation in
// VALU (hardware v_rcp/v_exp only — no IEEE divide chains), re-layout through
// LDS, then 1x v_wmma accumulating out = A @ G^T.
//
// Layouts (ISA 7.12.2, wave32):
//   A 16x32 f16 : lane L, M=L%16; K halves {0-7,16-23} (L<16) / {8-15,24-31}
//   B 32x16 f16 : lane L, N=L%16; this code uses K = 0..15 (L<16) / 16..31 (L>=16) for A and B
//   C/D 16x16 f32: lane L, vgpr r -> (M = r + 8*(L>=16), N = L%16)
// ---------------------------------------------------------------------------
__global__ void __launch_bounds__(32)
attn_out_kernel(const _Float16* __restrict__ Qh, const _Float16* __restrict__ Kh,
                const _Float16* __restrict__ Gh,
                const float* __restrict__ colmax, const float* __restrict__ colsum,
                const float* __restrict__ actw,
                float* __restrict__ out)
{
    __shared__ _Float16 ldsA[16 * 32];   // activated A tile, [row n][m-in-chunk]

    int bt   = blockIdx.x;
    int nb   = blockIdx.y * 16;
    int lane = threadIdx.x;
    int col  = lane & 15;
    bool lo  = lane < 16;
    int rowoff = lo ? 0 : 8;

    const _Float16* Qs = Qh + (size_t)bt * NN * NFP;
    const _Float16* Ks = Kh + (size_t)bt * NN * NFP;
    const _Float16* Gs = Gh + (size_t)bt * NFP * NN;
    const float* cmaxS = colmax + bt * NN;
    const float* csumS = colsum + bt * NN;

    float w0 = actw[0], w1 = actw[1], w2 = actw[2];

    // A operand (Q tile for rows nb..nb+15), fixed across the whole m loop.
    union { v16h v; v8h h[2]; } aQ;
    v8h zero8 = {};
    aQ.h[0] = lo ? *(const v8h*)(Qs + (nb + col) * NFP) : zero8;      // K 0-7 / 16-23 (pad)
    aQ.h[1] = lo ? *(const v8h*)(Qs + (nb + col) * NFP + 8) : zero8;  // K 8-15 / 24-31 (pad)

    v8f acc = {};   // out[n, f] accumulator

    for (int mb = 0; mb < NN; mb += 32) {
#pragma unroll
        for (int t = 0; t < 2; ++t) {
            int m0 = mb + t * 16;
            // B operand: K columns m0..m0+15
            v16h bk = {};
            if (lo) bk = *(const v16h*)(Ks + (m0 + col) * NFP);   // K 0-15; hi lanes = pad
            v8f s8 = {};
            s8 = __builtin_amdgcn_wmma_f32_16x16x32_f16(
                     false, aQ.v, false, bk, (short)0, s8, false, false);

            // every lane's 8 elements sit in column m = m0 + col
            float cm  = cmaxS[m0 + col];
            // w2 / colsum via hardware reciprocal (single v_rcp_f32)
            float inv = w2 * __builtin_amdgcn_rcpf(csumS[m0 + col]);
#pragma unroll
            for (int r = 0; r < 8; ++r) {
                float s = s8[r];
                // sigmoid via hardware rcp: w1 * rcp(1 + exp(-s))
                float sig = __builtin_amdgcn_rcpf(1.f + __expf(-s));
                float a = w0 * fmaxf(s, 0.f)
                        + w1 * sig
                        + inv * __expf(s - cm);
                ldsA[(r + rowoff) * 32 + t * 16 + col] = (_Float16)a;
            }
        }
        __syncthreads();

        // Re-load the activated tile as the A operand of the out-GEMM.
        union { v16h v; v8h h[2]; } aA;
        int k0 = lo ? 0 : 16;
        aA.h[0] = *(const v8h*)(&ldsA[col * 32 + k0]);        // K {0-7} / {16-23}
        aA.h[1] = *(const v8h*)(&ldsA[col * 32 + k0 + 8]);    // K {8-15} / {24-31}

        // B operand: G^T tile, N = f (lane), K = m rows of this chunk.
        v16h bG = *(const v16h*)(Gs + col * NN + mb + (lo ? 0 : 16));

        acc = __builtin_amdgcn_wmma_f32_16x16x32_f16(
                  false, aA.v, false, bG, (short)0, acc, false, false);
        __syncthreads();
    }

    // D layout -> out[bt][f][n], f = col (only f < 10 valid), n = nb + r + rowoff
    if (col < NF) {
#pragma unroll
        for (int r = 0; r < 8; ++r) {
            out[((size_t)bt * NF + col) * NN + nb + r + rowoff] = acc[r];
        }
    }
}

// ---------------------------------------------------------------------------
extern "C" void kernel_launch(void* const* d_in, const int* in_sizes, int n_in,
                              void* d_out, int out_size, void* d_ws, size_t ws_size,
                              hipStream_t stream)
{
    const float* input1 = (const float*)d_in[0];
    const float* input2 = (const float*)d_in[1];
    const float* actw   = (const float*)d_in[2];
    const float* W1 = (const float*)d_in[3];
    const float* b1 = (const float*)d_in[4];
    const float* W2 = (const float*)d_in[5];
    const float* b2 = (const float*)d_in[6];
    const float* W3 = (const float*)d_in[7];
    const float* b3 = (const float*)d_in[8];
    float* out = (float*)d_out;

    // workspace layout (≈3.3 MB, fits L2 trivially)
    char* ws = (char*)d_ws;
    _Float16* Qh = (_Float16*)(ws);                         // 32*1024*16*2 = 1 MiB
    _Float16* Kh = (_Float16*)(ws + (1u << 20));            // 1 MiB
    _Float16* Gh = (_Float16*)(ws + (2u << 20));            // 1 MiB
    float* colmax = (float*)(ws + (3u << 20));              // 128 KiB
    float* colsum = (float*)(ws + (3u << 20) + (128u << 10));

    proj_kernel<<<dim3(NBT, NN / 256), 256, 0, stream>>>(
        input1, input2, W1, b1, W2, b2, W3, b3, Qh, Kh, Gh);

    colstats_kernel<<<dim3(NBT, NN / 256), 256, 0, stream>>>(
        Qh, Kh, colmax, colsum);

    attn_out_kernel<<<dim3(NBT, NN / 16), 32, 0, stream>>>(
        Qh, Kh, Gh, colmax, colsum, actw, out);
}
